// GNN_12395275616823
// MI455X (gfx1250) — compile-verified, matrix operand order checked by reading, not run on hardware
//
#include <hip/hip_runtime.h>

// CDNA5 / gfx1250, wave32. WMMA types per CDNA5_HIP.md.
typedef __attribute__((ext_vector_type(16))) _Float16 v16h;
typedef __attribute__((ext_vector_type(8)))  float    v8f;

#define B_ 4
#define N_ 20000
#define E_ 320000
#define H_ 5
#define STEPS_ 10
#define ROWS_ (B_ * E_)            // 1,280,000 edge-rows (divisible by 16)
#define TILES_ (ROWS_ / 16)        // 80,000 row-tiles
#define WPB_ 8                     // waves per block (256 threads)
#define MSG_BLOCKS (TILES_ / WPB_) // 10,000 blocks
#define NFRAG_ 14                  // 4 (W1) + 8 (W2) + 2 (W3) weight fragments

// ---- WMMA fragment index helper (ISA 7.12.2, 16-bit A 16x32 layout) ----
// half-slot q (0..15) of v16h, hi = lane/16:
//   r = q/2, h = q%2 ; K = (r/4)*16 + (r%4)*2 + hi*8 + h
__device__ __forceinline__ int frag_k(int q, int hi) {
  int r = q >> 1, h = q & 1;
  return ((r >> 2) << 4) | ((r & 3) << 1) | (hi << 3) | h;
}

// Inverse permutation: column K -> storage slot so that a lane's 16 A-fragment
// halves are CONTIGUOUS.  slot = (chunk(K)<<5) | swap_bits_3_4(K%32):
// lane (M, hi) then reads halves [kbase + 16*hi .. +15] of row M in one b256 read.
__device__ __forceinline__ int a_slot(int K) {
  int c = K & 32;          // 32-wide K-chunk bit
  int k = K & 31;
  return c | (k & 7) | ((k >> 1) & 8) | ((k << 1) & 16);  // swap bits 3<->4
}

// A fragment: tile stored as 16 rows x 4 v16h (64 halves, permuted columns)
__device__ __forceinline__ v16h load_a(const v16h* T, int lane, int kc) {
  int M = lane & 15, hi = lane >> 4;
  return T[M * 4 + kc * 2 + hi];
}

// C init = bias broadcast down M (C layout: lane%16 -> N, vgpr -> M); bias in LDS
__device__ __forceinline__ v8f c_bias(const float* biasLds, int nbase, int lane) {
  float v = biasLds[nbase + (lane & 15)];
  v8f c;
#pragma unroll
  for (int i = 0; i < 8; ++i) c[i] = v;
  return c;
}

// relu(C) -> LDS activation tile in next-layer A-fragment order
__device__ __forceinline__ void store_c_relu(v16h* T, int nbase, v8f c, int lane) {
  _Float16* Th = (_Float16*)T;
  int hi = lane >> 4;
  int s = a_slot(nbase + (lane & 15));
#pragma unroll
  for (int i = 0; i < 8; ++i) {
    int M = i + 8 * hi;
    float v = c[i];
    Th[M * 64 + s] = (_Float16)(v > 0.0f ? v : 0.0f);
  }
}

// ---------------- edge message MLP + atomic scatter-add ----------------
// One wave per 16-row tile of the (B*E) edge-row space.
__global__ __launch_bounds__(256) void msg_kernel(
    const float* __restrict__ h_src, const float* __restrict__ bn,
    const float* __restrict__ J, const int* __restrict__ iin, const int* __restrict__ iout,
    const float* __restrict__ W1, const float* __restrict__ b1,
    const float* __restrict__ W2, const float* __restrict__ b2,
    const float* __restrict__ W3, const float* __restrict__ b3,
    float* __restrict__ mdst)
{
  __shared__ v16h  wfrag[NFRAG_ * 32];      // weight fragments, fragment-ordered (14 KB)
  __shared__ float blds[144];               // b1[0:64] b2[64:128] b3(pad16)[128:144]
  __shared__ v16h  abuf[WPB_][2][16 * 4];   // 2 activation tiles per wave (32 KB)

  const int tid  = threadIdx.x;
  const int wave = tid >> 5;
  const int lane = tid & 31;
  const int tile = blockIdx.x * WPB_ + wave;
  const int rowBase = tile * 16;

  // ---- one-time per-block staging: weights -> f16 fragments, biases -> LDS ----
  {
    _Float16* wh = (_Float16*)wfrag;
    for (int idx = tid; idx < NFRAG_ * 512; idx += 256) {
      int f = idx >> 9, rem = idx & 511, ln = rem >> 4, q = rem & 15;
      int hi = ln >> 4, n16 = ln & 15;
      const float* W;
      int ncols, nvalid, kmax, Nc, K;
      if (f < 4)       { W = W1; ncols = 64; nvalid = 64; kmax = 9;
                         Nc = f * 16 + n16;            K = frag_k(q, hi); }
      else if (f < 12) { W = W2; ncols = 64; nvalid = 64; kmax = 64; int g = f - 4;
                         Nc = (g >> 1) * 16 + n16;     K = ((g & 1) << 5) | frag_k(q, hi); }
      else             { W = W3; ncols = H_; nvalid = H_; kmax = 64;
                         Nc = n16;                     K = ((f - 12) << 5) | frag_k(q, hi); }
      float v = (K < kmax && Nc < nvalid) ? W[K * ncols + Nc] : 0.0f;
      wh[idx] = (_Float16)v;
    }
    if (tid < 64)       blds[tid] = b1[tid];
    else if (tid < 128) blds[tid] = b2[tid - 64];
    else if (tid < 144) blds[tid] = (tid - 128 < H_) ? b3[tid - 128] : 0.0f;
  }

  // hint the index slices for this tile into cache (global_prefetch_b8)
  if (lane == 0) {
    int e0 = rowBase % E_;
    __builtin_prefetch(iout + e0, 0, 1);
    __builtin_prefetch(iin + e0, 0, 1);
  }

  // ---- stage layer-1 input: 16 rows x 9 features (permuted cols), pad K<32 ----
  _Float16* a0h = (_Float16*)abuf[wave][0];
  for (int t = lane; t < 16 * 32; t += 32) {       // a_slot is a bijection on 0..31
    int M = t >> 5;
    a0h[M * 64 + (t & 31)] = (_Float16)0.0f;
  }
  if (lane < 16) {
    int g = rowBase + lane;
    if (g >= ROWS_) g = 0;                         // safety; exact fit in practice
    int bb = g / E_;
    int e  = g - bb * E_;
    int no = iout[e];
    int ni = iin[e];
    const float* hr = h_src + ((size_t)bb * N_ + no) * H_;
    float jv = J[(size_t)bb * E_ + e];
    _Float16* row = a0h + lane * 64;
#pragma unroll
    for (int j = 0; j < H_; ++j) row[j] = (_Float16)hr[j];   // a_slot(j)==j for j<8
    row[5]  = (_Float16)bn[(size_t)bb * N_ + ni];
    row[6]  = (_Float16)bn[(size_t)bb * N_ + no];
    row[7]  = (_Float16)jv;
    row[16] = (_Float16)(-jv);                               // a_slot(8) == 16
  }
  __syncthreads();

  const v16h* A0 = abuf[wave][0];
  v16h*       A1 = abuf[wave][1];

  // ---- layer 1: (16x9)@(9x64)+b1, relu -> A1 ----
  {
    v16h a = load_a(A0, lane, 0);
#pragma unroll
    for (int nt = 0; nt < 4; ++nt) {
      v8f c = c_bias(blds, nt * 16, lane);
      c = __builtin_amdgcn_wmma_f32_16x16x32_f16(false, a, false, wfrag[nt * 32 + lane],
                                                 (short)0, c, false, false);
      store_c_relu(A1, nt * 16, c, lane);
    }
  }
  __syncthreads();

  // ---- layer 2: (16x64)@(64x64)+b2, relu -> A0 (reuse) ----
  {
    v16h x0 = load_a(A1, lane, 0);
    v16h x1 = load_a(A1, lane, 1);
    v16h* A2 = abuf[wave][0];
#pragma unroll
    for (int nt = 0; nt < 4; ++nt) {
      v8f c = c_bias(blds + 64, nt * 16, lane);
      c = __builtin_amdgcn_wmma_f32_16x16x32_f16(false, x0, false,
                                                 wfrag[(4 + nt * 2) * 32 + lane],
                                                 (short)0, c, false, false);
      c = __builtin_amdgcn_wmma_f32_16x16x32_f16(false, x1, false,
                                                 wfrag[(5 + nt * 2) * 32 + lane],
                                                 (short)0, c, false, false);
      store_c_relu(A2, nt * 16, c, lane);
    }
  }
  __syncthreads();

  // ---- layer 3: (16x64)@(64x5)+b3, then atomic scatter-add into m ----
  {
    v16h x0 = load_a(A0, lane, 0);
    v16h x1 = load_a(A0, lane, 1);
    v8f c = c_bias(blds + 128, 0, lane);
    c = __builtin_amdgcn_wmma_f32_16x16x32_f16(false, x0, false, wfrag[12 * 32 + lane],
                                               (short)0, c, false, false);
    c = __builtin_amdgcn_wmma_f32_16x16x32_f16(false, x1, false, wfrag[13 * 32 + lane],
                                               (short)0, c, false, false);

    int hi = lane >> 4, Nc = lane & 15;
    if (Nc < H_) {                         // divergence only AFTER all WMMAs
#pragma unroll
      for (int i = 0; i < 8; ++i) {
        int M = i + 8 * hi;
        int g = rowBase + M;
        if (g < ROWS_) {
          int bb = g / E_;
          int e  = g - bb * E_;
          int no = iout[e];
          atomicAdd(&mdst[((size_t)bb * N_ + no) * H_ + Nc], c[i]);
        }
      }
    }
  }
}

// ---------------- per-node GRU update (tiny: scalar VALU) ----------------
__global__ void gru_kernel(const float* __restrict__ m0, const float* __restrict__ m1,
                           float* __restrict__ h0, float* __restrict__ h1,
                           const float* __restrict__ gzW, const float* __restrict__ gzb,
                           const float* __restrict__ grW, const float* __restrict__ grb,
                           const float* __restrict__ ghW, const float* __restrict__ ghb)
{
  int i = blockIdx.x * blockDim.x + threadIdx.x;
  if (i >= B_ * N_) return;
  float a[15];
#pragma unroll
  for (int j = 0; j < H_; ++j) {
    a[j]      = h1[(size_t)i * H_ + j];
    a[5 + j]  = m0[(size_t)i * H_ + j];
    a[10 + j] = m1[(size_t)i * H_ + j];
  }
  float z[H_], r[H_];
#pragma unroll
  for (int o = 0; o < H_; ++o) {
    float sz = gzb[o], sr = grb[o];
#pragma unroll
    for (int k = 0; k < 15; ++k) { sz += a[k] * gzW[k * H_ + o]; sr += a[k] * grW[k * H_ + o]; }
    z[o] = 1.0f / (1.0f + __expf(-sz));
    r[o] = 1.0f / (1.0f + __expf(-sr));
  }
  float jn[15];
#pragma unroll
  for (int k = 0; k < 15; ++k) jn[k] = (k < H_) ? r[k] * a[k] : a[k];
  float hh[H_];
#pragma unroll
  for (int o = 0; o < H_; ++o) {
    float s = ghb[o];
#pragma unroll
    for (int k = 0; k < 15; ++k) s += jn[k] * ghW[k * H_ + o];
    hh[o] = tanhf(s);
  }
#pragma unroll
  for (int o = 0; o < H_; ++o) {
    float old = a[o];
    h0[(size_t)i * H_ + o] = old;                                // h_t_0 = h_t_1
    h1[(size_t)i * H_ + o] = (1.0f - z[o]) * old + z[o] * hh[o]; // h_t_1 = h_new
  }
}

// ---------------- per-node output MLP (tiny: scalar VALU) ----------------
__global__ void out_kernel(const float* __restrict__ h1, const float* __restrict__ bn,
                           const float* __restrict__ oW1, const float* __restrict__ ob1,
                           const float* __restrict__ oW2, const float* __restrict__ ob2,
                           const float* __restrict__ oW3, const float* __restrict__ ob3,
                           float* __restrict__ out)
{
  int i = blockIdx.x * blockDim.x + threadIdx.x;
  if (i >= B_ * N_) return;
  float x[6];
#pragma unroll
  for (int j = 0; j < H_; ++j) x[j] = h1[(size_t)i * H_ + j];
  x[5] = bn[i];
  float t1[64];
#pragma unroll
  for (int o = 0; o < 64; ++o) {
    float s = ob1[o];
#pragma unroll
    for (int k = 0; k < 6; ++k) s += x[k] * oW1[k * 64 + o];
    t1[o] = s > 0.0f ? s : 0.0f;
  }
  float t2[64];
  for (int o = 0; o < 64; ++o) {
    float s = ob2[o];
    for (int k = 0; k < 64; ++k) s += t1[k] * oW2[k * 64 + o];
    t2[o] = s > 0.0f ? s : 0.0f;
  }
  float s = ob3[0];
  for (int k = 0; k < 64; ++k) s += t2[k] * oW3[k];
  out[i] = 1.0f / (1.0f + __expf(-s));
}

__global__ void zero_kernel(float* __restrict__ p, int n) {
  int i = blockIdx.x * blockDim.x + threadIdx.x;
  if (i < n) p[i] = 0.0f;
}

extern "C" void kernel_launch(void* const* d_in, const int* in_sizes, int n_in,
                              void* d_out, int out_size, void* d_ws, size_t ws_size,
                              hipStream_t stream)
{
  const float* J   = (const float*)d_in[0];
  const float* bn  = (const float*)d_in[1];
  const int*   iin = (const int*)d_in[2];
  const int*   iout= (const int*)d_in[3];
  const float* mW1 = (const float*)d_in[4];  const float* mb1 = (const float*)d_in[5];
  const float* mW2 = (const float*)d_in[6];  const float* mb2 = (const float*)d_in[7];
  const float* mW3 = (const float*)d_in[8];  const float* mb3 = (const float*)d_in[9];
  const float* gzW = (const float*)d_in[10]; const float* gzb = (const float*)d_in[11];
  const float* grW = (const float*)d_in[12]; const float* grb = (const float*)d_in[13];
  const float* ghW = (const float*)d_in[14]; const float* ghb = (const float*)d_in[15];
  const float* oW1 = (const float*)d_in[16]; const float* ob1 = (const float*)d_in[17];
  const float* oW2 = (const float*)d_in[18]; const float* ob2 = (const float*)d_in[19];
  const float* oW3 = (const float*)d_in[20]; const float* ob3 = (const float*)d_in[21];

  float* ws = (float*)d_ws;
  const size_t S = (size_t)B_ * N_ * H_;   // 400,000 floats per state buffer
  float* h0 = ws;
  float* h1 = ws + S;
  float* m0 = ws + 2 * S;
  float* m1 = ws + 3 * S;

  // deterministic zero-init of all state every call
  int zn = (int)(4 * S);
  zero_kernel<<<(zn + 255) / 256, 256, 0, stream>>>(ws, zn);

  const int nodeThreads = B_ * N_;
  const int nodeBlocks  = (nodeThreads + 255) / 256;

  for (int step = 0; step < STEPS_; ++step) {
    // m_t_0 += scatter(message(h_t_0[index_out]))
    msg_kernel<<<MSG_BLOCKS, 256, 0, stream>>>(h0, bn, J, iin, iout,
                                               mW1, mb1, mW2, mb2, mW3, mb3, m0);
    // m_t_1 = (updated m_t_0) ...
    hipMemcpyAsync(m1, m0, S * sizeof(float), hipMemcpyDeviceToDevice, stream);
    // ... += scatter(message(h_t_1[index_out]))
    msg_kernel<<<MSG_BLOCKS, 256, 0, stream>>>(h1, bn, J, iin, iout,
                                               mW1, mb1, mW2, mb2, mW3, mb3, m1);
    // GRU node update (h0 <- h1, h1 <- h_new)
    gru_kernel<<<nodeBlocks, 256, 0, stream>>>(m0, m1, h0, h1,
                                               gzW, gzb, grW, grb, ghW, ghb);
  }
  out_kernel<<<nodeBlocks, 256, 0, stream>>>(h1, bn, oW1, ob1, oW2, ob2, oW3, ob3,
                                             (float*)d_out);
}
